// global_Interp_3874060501425
// MI455X (gfx1250) — matrix operand order repaired and checked by reading, not compiled, not run on hardware
//
#include <hip/hip_runtime.h>
#include <hip/hip_bf16.h>

typedef __attribute__((ext_vector_type(16))) _Float16 v16h;
typedef __attribute__((ext_vector_type(8)))  _Float16 v8h;
typedef __attribute__((ext_vector_type(8)))  float    v8f;

#define T_DIM 4
#define N_DIM 256
#define D_DIM 128
#define NT    (T_DIM * N_DIM)   // 1024 rows total
#define BN_EPS 1e-5f
#define NEG_SLOPE 0.01f

// ---------------------------------------------------------------- helpers

// leaky relu as fmax(x, 0.01x): identical result, avoids select->fabs combines
__device__ __forceinline__ float lrelu(float x) { return fmaxf(x, NEG_SLOPE * x); }

// scalar |x| via explicit sign-mask in inline asm: opaque to the SLP vectorizer,
// prevents the illegal v2f32-fabs combine that crashes the gfx1250 backend.
__device__ __forceinline__ float fabs_asm(float x) {
    float r;
    asm("v_and_b32 %0, 0x7fffffff, %1" : "=v"(r) : "v"(x));
    return r;
}

__device__ __forceinline__ v8f wmma16(v16h a, v16h b, v8f c) {
    return __builtin_amdgcn_wmma_f32_16x16x32_f16(
        /*neg_a=*/false, a, /*neg_b=*/false, b,
        /*c_mod=*/(short)0, c, /*reuse_a=*/false, /*reuse_b=*/false);
}

// A-matrix fragment (16x32 f16): lane holds row (lane&15),
// K = {k0..k0+7, k0+16..k0+23}, k0 = kc*32 + (lane>>4)*8.  Row-major src [row][ld].
__device__ __forceinline__ v16h ld_frag_a(const _Float16* __restrict__ base,
                                          int row, int ld, int kc, int laneHi) {
    int k0 = kc * 32 + laneHi * 8;
    const v8h lo = *(const v8h*)(base + row * ld + k0);
    const v8h hi = *(const v8h*)(base + row * ld + k0 + 16);
    return __builtin_shufflevector(lo, hi, 0,1,2,3,4,5,6,7,8,9,10,11,12,13,14,15);
}

// B-matrix fragment (32x16 f16): lane holds col (lane&15),
// K = {k0..k0+7, k0+16..k0+23}, k0 = kc*32 + (lane>>4)*8.  Weights stored row-major [col(c)][K(d)].
__device__ __forceinline__ v16h ld_frag_b(const _Float16* __restrict__ base,
                                          int col, int ld, int kc, int laneHi) {
    int k0 = kc * 32 + laneHi * 8;
    const v8h lo = *(const v8h*)(base + col * ld + k0);
    const v8h hi = *(const v8h*)(base + col * ld + k0 + 16);
    return __builtin_shufflevector(lo, hi, 0,1,2,3,4,5,6,7,8,9,10,11,12,13,14,15);
}

// ---------------------------------------------------------------- prep: fold BN into weights

__global__ __launch_bounds__(256) void prep_fold(
    const float* __restrict__ w, const float* __restrict__ g,
    const float* __restrict__ b, const float* __restrict__ m,
    const float* __restrict__ v, _Float16* __restrict__ wf,
    float* __restrict__ tf, int C, int K)
{
    int idx = blockIdx.x * 256 + threadIdx.x;
    if (idx >= C * K) return;
    int c = idx / K, k = idx % K;
    float s = g[c] * rsqrtf(v[c] + BN_EPS);
    wf[idx] = (_Float16)(w[idx] * s);
    if (k == 0) tf[c] = b[c] - m[c] * s;
}

// ---------------------------------------------------------------- fused edge MLP
// One block (8 waves) per (t, i).  Each wave processes 2 groups of 16 j rows.
// GEMM1: X[16x128] @ W0'[128->256], GEMM2: H0[16x256] @ W1'[256->128],
// out: sigmoid(H1 @ wout + bout) + eye + 1e-6; atomic column sums for axis-1 norm.
// Fragment fetches are double-buffered so ds_loads overlap WMMA execution.

__global__ __launch_bounds__(256) void edge_mlp_kernel(
    const float* __restrict__ node,
    const _Float16* __restrict__ gW0, const float* __restrict__ gT0,
    const _Float16* __restrict__ gW1, const float* __restrict__ gT1,
    const float* __restrict__ gWout, const float* __restrict__ gBout,
    float* __restrict__ efp, float* __restrict__ colsum)
{
    extern __shared__ __align__(16) char smem[];
    _Float16* sW0 = (_Float16*)smem;            // 256*128 f16
    _Float16* sW1 = sW0 + 256 * 128;            // 128*256 f16
    float*    sT0 = (float*)(sW1 + 128 * 256);  // 256 f32
    float*    sT1 = sT0 + 256;                  // 128 f32
    float*    sWo = sT1 + 128;                  // 128 f32
    _Float16* sH0 = (_Float16*)(sWo + 128);     // 8 waves * 16 * 256 f16

    const int tid  = threadIdx.x;
    const int bid  = blockIdx.x;
    const int t    = bid >> 8;
    const int i    = bid & 255;

    // stage folded weights into LDS
    {
        const uint4* s0 = (const uint4*)gW0; uint4* d0 = (uint4*)sW0;
        const uint4* s1 = (const uint4*)gW1; uint4* d1 = (uint4*)sW1;
        #pragma unroll
        for (int r = 0; r < 16; ++r) {
            d0[tid + r * 256] = s0[tid + r * 256];
            d1[tid + r * 256] = s1[tid + r * 256];
        }
        sT0[tid] = gT0[tid];
        if (tid < 128) { sT1[tid] = gT1[tid]; sWo[tid] = gWout[tid]; }
    }
    __syncthreads();

    const int wave   = tid >> 5;
    const int lane   = tid & 31;
    const int laneM  = lane & 15;
    const int laneHi = lane >> 4;
    const float bout = gBout[0];

    const float* nrow_i = node + (size_t)(t * N_DIM + i) * D_DIM;
    _Float16* h0w = sH0 + wave * 16 * 256;

    for (int jg = 0; jg < 2; ++jg) {
        const int jb = (jg * 8 + wave) * 16;
        const int j  = jb + laneM;
        const float* nrow_j = node + (size_t)(t * N_DIM + j) * D_DIM;

        // ---- build X = |node_i - node_j| fragments (16x128 as 4 k-chunks)
        v16h xf[4];
        #pragma unroll
        for (int kc = 0; kc < 4; ++kc) {
            int k0 = kc * 32 + laneHi * 8;
            union { float4 v4[4]; float f[16]; } fi, fj;
            fi.v4[0] = *(const float4*)(nrow_i + k0);
            fi.v4[1] = *(const float4*)(nrow_i + k0 + 4);
            fi.v4[2] = *(const float4*)(nrow_i + k0 + 16);
            fi.v4[3] = *(const float4*)(nrow_i + k0 + 20);
            fj.v4[0] = *(const float4*)(nrow_j + k0);
            fj.v4[1] = *(const float4*)(nrow_j + k0 + 4);
            fj.v4[2] = *(const float4*)(nrow_j + k0 + 16);
            fj.v4[3] = *(const float4*)(nrow_j + k0 + 20);
            v16h x;
            #pragma unroll
            for (int e = 0; e < 16; ++e)
                x[e] = (_Float16)fabs_asm(fi.f[e] - fj.f[e]);
            xf[kc] = x;
        }

        // ---- GEMM1: H0 = leaky(X @ W0' + t0); double-buffered B fragments
        {
            v16h bf0[4], bf1[4];
            #pragma unroll
            for (int kc = 0; kc < 4; ++kc)
                bf0[kc] = ld_frag_b(sW0, laneM, 128, kc, laneHi);   // ct = 0
            #pragma unroll
            for (int ct = 0; ct < 16; ++ct) {
                v16h (&cur)[4] = (ct & 1) ? bf1 : bf0;
                v16h (&nxt)[4] = (ct & 1) ? bf0 : bf1;
                if (ct < 15) {
                    #pragma unroll
                    for (int kc = 0; kc < 4; ++kc)
                        nxt[kc] = ld_frag_b(sW0, (ct + 1) * 16 + laneM, 128, kc, laneHi);
                }
                v8f acc = {0.f,0.f,0.f,0.f,0.f,0.f,0.f,0.f};
                #pragma unroll
                for (int kc = 0; kc < 4; ++kc)
                    acc = wmma16(xf[kc], cur[kc], acc);
                int c = ct * 16 + laneM;
                float bv = sT0[c];
                #pragma unroll
                for (int e = 0; e < 8; ++e) {
                    float vv = lrelu(acc[e] + bv);
                    h0w[(e + 8 * laneHi) * 256 + c] = (_Float16)vv;
                }
            }
        }

        // ---- GEMM2: H1 = leaky(H0 @ W1' + t1); fold in output dot with wout.
        // Flattened (ct,kc) loop, A and B fragments double-buffered one step ahead.
        float sp[8] = {0.f,0.f,0.f,0.f,0.f,0.f,0.f,0.f};
        {
            v16h afb[2], bfb[2];
            afb[0] = ld_frag_a(h0w, laneM, 256, 0, laneHi);
            bfb[0] = ld_frag_b(sW1, laneM, 256, 0, laneHi);
            v8f acc = {0.f,0.f,0.f,0.f,0.f,0.f,0.f,0.f};
            #pragma unroll
            for (int it = 0; it < 64; ++it) {
                const int ct = it >> 3, kc = it & 7;
                const int nit = it + 1;
                if (nit < 64) {
                    afb[nit & 1] = ld_frag_a(h0w, laneM, 256, nit & 7, laneHi);
                    bfb[nit & 1] = ld_frag_b(sW1, (nit >> 3) * 16 + laneM, 256, nit & 7, laneHi);
                }
                acc = wmma16(afb[it & 1], bfb[it & 1], acc);
                if (kc == 7) {
                    int c = ct * 16 + laneM;
                    float bv = sT1[c], wo = sWo[c];
                    #pragma unroll
                    for (int e = 0; e < 8; ++e) sp[e] += lrelu(acc[e] + bv) * wo;
                    acc = v8f{0.f,0.f,0.f,0.f,0.f,0.f,0.f,0.f};
                }
            }
        }

        // reduce the 128-wide dot across the 16 lanes of each half-wave
        #pragma unroll
        for (int e = 0; e < 8; ++e) {
            sp[e] += __shfl_xor(sp[e], 1, 16);
            sp[e] += __shfl_xor(sp[e], 2, 16);
            sp[e] += __shfl_xor(sp[e], 4, 16);
            sp[e] += __shfl_xor(sp[e], 8, 16);
        }

        if (laneM == 0) {
            #pragma unroll
            for (int e = 0; e < 8; ++e) {
                int jj = jb + e + 8 * laneHi;
                float sg  = 1.f / (1.f + __expf(-(sp[e] + bout)));
                float efv = sg + ((jj == i) ? 1.f : 0.f) + 1e-6f;
                efp[(size_t)(t * N_DIM + i) * N_DIM + jj] = efv;
                atomicAdd(&colsum[t * N_DIM + jj], efv);
            }
        }
    }
}

// ---------------------------------------------------------------- normalize ef, aggregate, pack [node||aggr] f16
// One wave per (t,i).

__global__ __launch_bounds__(256) void aggr_kernel(
    const float* __restrict__ efp, const float* __restrict__ colsum,
    const float* __restrict__ node, _Float16* __restrict__ hcat,
    float* __restrict__ ef_out)
{
    const int wave = threadIdx.x >> 5;
    const int lane = threadIdx.x & 31;
    const int wid  = blockIdx.x * 8 + wave;
    const int t = wid >> 8, i = wid & 255;

    const float* efr = efp + (size_t)(t * N_DIM + i) * N_DIM;
    float* efo       = ef_out + (size_t)(t * N_DIM + i) * N_DIM;

    float em[8];
    float rs = 0.f;
    #pragma unroll
    for (int q = 0; q < 8; ++q) {
        int j = q * 32 + lane;
        float v = efr[j] / colsum[t * N_DIM + j];   // normalize over axis=1
        efo[j] = v;                                  // final ef output (last layer wins)
        float e = (j == i) ? 0.f : v;                // mask diagonal
        em[q] = e;
        rs += e;
    }
    rs += __shfl_xor(rs, 1, 32);  rs += __shfl_xor(rs, 2, 32);
    rs += __shfl_xor(rs, 4, 32);  rs += __shfl_xor(rs, 8, 32);
    rs += __shfl_xor(rs, 16, 32);
    const float inv = 1.f / fmaxf(rs, 1e-12f);

    float acc[4] = {0.f, 0.f, 0.f, 0.f};
    #pragma unroll
    for (int q = 0; q < 8; ++q) {
        float ev = em[q];
        for (int r = 0; r < 32; ++r) {
            float w = __shfl(ev, r, 32);
            const float* nr = node + (size_t)(t * N_DIM + q * 32 + r) * D_DIM;
            #pragma unroll
            for (int d4 = 0; d4 < 4; ++d4) acc[d4] += w * nr[lane + 32 * d4];
        }
    }

    const float* ni = node + (size_t)(t * N_DIM + i) * D_DIM;
    _Float16* hr = hcat + (size_t)(t * N_DIM + i) * 256;
    #pragma unroll
    for (int d4 = 0; d4 < 4; ++d4) {
        int d = lane + 32 * d4;
        hr[d]       = (_Float16)ni[d];
        hr[128 + d] = (_Float16)(acc[d4] * inv);
    }
}

// ---------------------------------------------------------------- node MLP layer 0: [1024,256] @ [256->256], f16 out

__global__ __launch_bounds__(256) void node_l0(
    const _Float16* __restrict__ A, const _Float16* __restrict__ W,
    const float* __restrict__ bias, _Float16* __restrict__ out)
{
    const int wave = threadIdx.x >> 5, lane = threadIdx.x & 31;
    const int laneM = lane & 15, laneHi = lane >> 4;
    const int wid = blockIdx.x * 8 + wave;
    const int rt = wid >> 4, ct = wid & 15;
    const int rowbase = rt * 16;

    v8f acc = {0.f,0.f,0.f,0.f,0.f,0.f,0.f,0.f};
    v16h a0 = ld_frag_a(A, rowbase + laneM, 256, 0, laneHi);
    v16h b0 = ld_frag_b(W, ct * 16 + laneM, 256, 0, laneHi);
    #pragma unroll
    for (int kc = 0; kc < 8; ++kc) {
        v16h a1, b1;
        if (kc < 7) {
            a1 = ld_frag_a(A, rowbase + laneM, 256, kc + 1, laneHi);
            b1 = ld_frag_b(W, ct * 16 + laneM, 256, kc + 1, laneHi);
        }
        acc = wmma16(a0, b0, acc);
        a0 = a1; b0 = b1;
    }

    int c = ct * 16 + laneM;
    float bv = bias[c];
    #pragma unroll
    for (int e = 0; e < 8; ++e)
        out[(size_t)(rowbase + e + 8 * laneHi) * 256 + c] = (_Float16)lrelu(acc[e] + bv);
}

// ---------------------------------------------------------------- node MLP layer 1: [1024,256] @ [256->128], f32 out

__global__ __launch_bounds__(256) void node_l1(
    const _Float16* __restrict__ A, const _Float16* __restrict__ W,
    const float* __restrict__ bias, float* __restrict__ out)
{
    const int wave = threadIdx.x >> 5, lane = threadIdx.x & 31;
    const int laneM = lane & 15, laneHi = lane >> 4;
    const int wid = blockIdx.x * 8 + wave;
    const int rt = wid >> 3, ct = wid & 7;
    const int rowbase = rt * 16;

    v8f acc = {0.f,0.f,0.f,0.f,0.f,0.f,0.f,0.f};
    v16h a0 = ld_frag_a(A, rowbase + laneM, 256, 0, laneHi);
    v16h b0 = ld_frag_b(W, ct * 16 + laneM, 256, 0, laneHi);
    #pragma unroll
    for (int kc = 0; kc < 8; ++kc) {
        v16h a1, b1;
        if (kc < 7) {
            a1 = ld_frag_a(A, rowbase + laneM, 256, kc + 1, laneHi);
            b1 = ld_frag_b(W, ct * 16 + laneM, 256, kc + 1, laneHi);
        }
        acc = wmma16(a0, b0, acc);
        a0 = a1; b0 = b1;
    }

    int c = ct * 16 + laneM;
    float bv = bias[c];
    #pragma unroll
    for (int e = 0; e < 8; ++e)
        out[(size_t)(rowbase + e + 8 * laneHi) * 128 + c] = lrelu(acc[e] + bv);
}

// ---------------------------------------------------------------- launch

extern "C" void kernel_launch(void* const* d_in, const int* in_sizes, int n_in,
                              void* d_out, int out_size, void* d_ws, size_t ws_size,
                              hipStream_t stream) {
    (void)in_sizes; (void)n_in; (void)out_size; (void)ws_size;

    const float* node_feat = (const float*)d_in[0];
    // d_in[1] = edge_feat (unused by the net)

    // pytree leaf order: layers list, then sorted dict keys
    // edge: bn0{b,g,m,v}, bn1{b,g,m,v}, bout, w0, w1, wout ; node: bn0{..}, bn1{..}, w0, w1
    auto P = [&](int l, int k) -> const float* { return (const float*)d_in[2 + 22 * l + k]; };

    // workspace carve
    char* ws = (char*)d_ws;
    size_t off = 0;
    float*    efp     = (float*)(ws + off); off += (size_t)T_DIM * N_DIM * N_DIM * 4;
    float*    colsum  = (float*)(ws + off); off += (size_t)T_DIM * N_DIM * 4;
    _Float16* hcat    = (_Float16*)(ws + off); off += (size_t)NT * 256 * 2;
    _Float16* h0n     = (_Float16*)(ws + off); off += (size_t)NT * 256 * 2;
    float*    nodebuf = (float*)(ws + off); off += (size_t)NT * D_DIM * 4;

    _Float16 *eW0[2], *eW1[2], *nW0[2], *nW1[2];
    float *et0[2], *et1[2], *nt0[2], *nt1[2];
    for (int l = 0; l < 2; ++l) {
        eW0[l] = (_Float16*)(ws + off); off += 256 * 128 * 2;
        eW1[l] = (_Float16*)(ws + off); off += 128 * 256 * 2;
        nW0[l] = (_Float16*)(ws + off); off += 256 * 256 * 2;
        nW1[l] = (_Float16*)(ws + off); off += 128 * 256 * 2;
        et0[l] = (float*)(ws + off); off += 256 * 4;
        et1[l] = (float*)(ws + off); off += 128 * 4;
        nt0[l] = (float*)(ws + off); off += 256 * 4;
        nt1[l] = (float*)(ws + off); off += 128 * 4;
    }

    // fold BN into weights (f16) for both layers
    for (int l = 0; l < 2; ++l) {
        prep_fold<<<(256 * 128 + 255) / 256, 256, 0, stream>>>(
            P(l, 9),  P(l, 1),  P(l, 0),  P(l, 2),  P(l, 3),  eW0[l], et0[l], 256, 128);
        prep_fold<<<(128 * 256 + 255) / 256, 256, 0, stream>>>(
            P(l, 10), P(l, 5),  P(l, 4),  P(l, 6),  P(l, 7),  eW1[l], et1[l], 128, 256);
        prep_fold<<<(256 * 256 + 255) / 256, 256, 0, stream>>>(
            P(l, 20), P(l, 13), P(l, 12), P(l, 14), P(l, 15), nW0[l], nt0[l], 256, 256);
        prep_fold<<<(128 * 256 + 255) / 256, 256, 0, stream>>>(
            P(l, 21), P(l, 17), P(l, 16), P(l, 18), P(l, 19), nW1[l], nt1[l], 128, 256);
    }

    float* out_node = (float*)d_out;                      // [T,N,128]
    float* out_ef   = (float*)d_out + (size_t)NT * D_DIM; // [T,N,N]

    const size_t smem = (256 * 128 + 128 * 256 + 8 * 16 * 256) * sizeof(_Float16)
                      + (256 + 128 + 128) * sizeof(float);   // ~198.6 KB dynamic LDS

    const float* node_cur = node_feat;
    for (int l = 0; l < 2; ++l) {
        (void)hipMemsetAsync(colsum, 0, (size_t)T_DIM * N_DIM * 4, stream);
        edge_mlp_kernel<<<T_DIM * N_DIM, 256, smem, stream>>>(
            node_cur, eW0[l], et0[l], eW1[l], et1[l], P(l, 11), P(l, 8), efp, colsum);
        aggr_kernel<<<NT / 8, 256, 0, stream>>>(efp, colsum, node_cur, hcat, out_ef);
        node_l0<<<(NT / 16) * 16 / 8, 256, 0, stream>>>(hcat, nW0[l], nt0[l], h0n);
        float* node_next = (l == 1) ? out_node : nodebuf;
        node_l1<<<(NT / 16) * 8 / 8, 256, 0, stream>>>(h0n, nW1[l], nt1[l], node_next);
        node_cur = nodebuf;
    }
}
